// MultiHead_34084860461312
// MI455X (gfx1250) — matrix operand order repaired and checked by reading, not compiled, not run on hardware
//
#include <hip/hip_runtime.h>

// ---------------------------------------------------------------------------
// Multi-head GRU scan (V=16 heads, B=64, T=256, H=128) + projection + FC.
//
// MI455X / gfx1250 (wave32) strategy, v2:
//  * grid = 16 heads x 4 batch tiles; block = 256 threads = 8 waves.
//  * Whh fragments (fp16) are LOOP-INVARIANT -> held in 96 VGPRs per wave,
//    converted f32->f16 directly from global at startup (no LDS staging).
//  * h kept in registers per lane (each lane re-reads exactly what it wrote);
//    only the fp16 copy of h goes to LDS (A-operand needed by all waves),
//    double-buffered -> single __syncthreads per timestep.
//  * x transposed in LDS to [T][16]: one contiguous 32B load per lane per step.
//  * Per step: 12 x v_wmma_f32_16x16x32_f16 per wave; wave w owns gate tiles
//    {w, 8+w, 16+w} = r/z/n of hidden slice [16w,16w+16) -> GRU nonlinearity
//    entirely in registers.
// ---------------------------------------------------------------------------

typedef _Float16 half8  __attribute__((ext_vector_type(8)));
typedef _Float16 half16 __attribute__((ext_vector_type(16)));
typedef float    v8f    __attribute__((ext_vector_type(8)));
typedef float    f4v    __attribute__((ext_vector_type(4)));

#define NV    16     // heads
#define NB    64     // batch
#define NT    256    // timesteps
#define NH    128    // hidden
#define G3    384    // 3*H gates
#define NFC   256
#define NC    10
#define BTILE 16

// A fragment (16-bit A 16x32): lane<16 gets K = k0+0..15,
// lane>=16 gets K = k0+16..31  (sub = lane>>4).
__device__ __forceinline__ half16 fragA_ld(const _Float16* base,
                                           int row, int k0, int sub) {
  const _Float16* p = base + row * NH + k0 + sub * 16;
  half8 lo = *(const half8*)(p);
  half8 hi = *(const half8*)(p + 8);
  return __builtin_shufflevector(lo, hi, 0,1,2,3,4,5,6,7,8,9,10,11,12,13,14,15);
}

__device__ __forceinline__ float sigmoidf_(float x) {
  return 1.0f / (1.0f + __expf(-x));
}
__device__ __forceinline__ float tanhf_(float x) {
  float e = __expf(-2.0f * x);          // tanh(x) = (1 - e^-2x) / (1 + e^-2x)
  return (1.0f - e) / (1.0f + e);
}

__global__ void __launch_bounds__(256)
gru_scan_kernel(const float* __restrict__ x,     // [V][B][T]
                const float* __restrict__ Wih,   // [V][384]
                const float* __restrict__ Whh,   // [V][384][128]
                const float* __restrict__ bih,   // [V][384]
                const float* __restrict__ bhh,   // [V][384]
                const float* __restrict__ Wp,    // [V][128]
                const float* __restrict__ bp,    // [V]
                float* __restrict__ concat)      // [B][V]
{
  __shared__ __align__(16) _Float16 h16[2 * BTILE * NH];  // 8 KB, double-buffered
  __shared__ __align__(16) float    h32[BTILE * NH];      // 8 KB (final h only)
  __shared__ __align__(16) float    xt[NT * BTILE];       // 16 KB, [t][m]

  const int v    = blockIdx.x;          // head
  const int b0   = blockIdx.y * BTILE;  // batch tile base
  const int tid  = threadIdx.x;
  const int lane = tid & 31;
  const int wv   = tid >> 5;            // wave 0..7
  const int sub  = (lane >> 4) & 1;
  const int col  = lane & 15;
  const int j    = wv * 16 + col;       // hidden column this lane updates

  // ---- B fragments (Whh^T), loop-invariant, straight from global to VGPRs.
  // B[k][n] = Whh[n][k]; lane holds N = col of tile, K = k0 + sub*16 + 0..15
  // (contiguous along Whh's k axis).
  const float* whh_g = Whh + v * G3 * NH;
  half16 Bf[3][4];                      // [gate r/z/n][k-chunk], 96 VGPRs
  #pragma unroll
  for (int g = 0; g < 3; ++g) {
    const float* rowp = whh_g + (g * NH + wv * 16 + col) * NH + sub * 16;
    #pragma unroll
    for (int kk = 0; kk < 4; ++kk) {
      const float* p = rowp + kk * 32;
      half16 hb;
      #pragma unroll
      for (int q = 0; q < 4; ++q) {
        f4v w = *(const f4v*)(p + q * 4);
        hb[q * 4 + 0] = (_Float16)w[0];
        hb[q * 4 + 1] = (_Float16)w[1];
        hb[q * 4 + 2] = (_Float16)w[2];
        hb[q * 4 + 3] = (_Float16)w[3];
      }
      Bf[g][kk] = hb;
    }
  }

  // ---- per-lane gate coefficients (gi = x*Wih + bih; gh bias = bhh)
  const int nr = wv * 16 + col, nz = NH + wv * 16 + col, nn = 2 * NH + wv * 16 + col;
  const float wir = Wih[v * G3 + nr], wiz = Wih[v * G3 + nz], win = Wih[v * G3 + nn];
  const float bir = bih[v * G3 + nr], biz = bih[v * G3 + nz], bin = bih[v * G3 + nn];
  const float bhr = bhh[v * G3 + nr], bhz = bhh[v * G3 + nz], bhn = bhh[v * G3 + nn];

  // ---- x preload, transposed to [t][m] so a lane's 8 rows are contiguous
  {
    int m  = tid & 15;
    int tb = (tid >> 4) * 16;
    const float* xr = x + (size_t)(v * NB + b0 + m) * NT + tb;
    #pragma unroll
    for (int i = 0; i < 16; ++i) xt[(tb + i) * BTILE + m] = xr[i];
  }
  // ---- h0 = 0 (both fp16 buffers) ----
  for (int i = tid; i < 2 * BTILE * NH; i += 256) h16[i] = (_Float16)0.0f;

  float hp[8];
  #pragma unroll
  for (int i = 0; i < 8; ++i) hp[i] = 0.0f;

  __syncthreads();

  #pragma unroll 1
  for (int t = 0; t < NT; ++t) {
    const _Float16* hc = h16 + (t & 1) * (BTILE * NH);        // read buffer
    _Float16*       hn = h16 + ((t + 1) & 1) * (BTILE * NH);  // write buffer

    half16 a[4];
    #pragma unroll
    for (int kk = 0; kk < 4; ++kk) a[kk] = fragA_ld(hc, col, kk * 32, sub);

    // x values for this lane's 8 rows (m = sub*8 + i): contiguous 32B
    f4v xa = *(const f4v*)(xt + t * BTILE + sub * 8);
    f4v xb = *(const f4v*)(xt + t * BTILE + sub * 8 + 4);

    v8f cr = {}, cz = {}, cn = {};
    #pragma unroll
    for (int kk = 0; kk < 4; ++kk) {
      cr = __builtin_amdgcn_wmma_f32_16x16x32_f16(false, a[kk], false, Bf[0][kk],
                                                  (short)0, cr, false, false);
      cz = __builtin_amdgcn_wmma_f32_16x16x32_f16(false, a[kk], false, Bf[1][kk],
                                                  (short)0, cz, false, false);
      cn = __builtin_amdgcn_wmma_f32_16x16x32_f16(false, a[kk], false, Bf[2][kk],
                                                  (short)0, cn, false, false);
    }

    #pragma unroll
    for (int i = 0; i < 8; ++i) {
      int m    = sub * 8 + i;
      float xv = (i < 4) ? xa[i] : xb[i - 4];
      float r  = sigmoidf_(xv * wir + bir + cr[i] + bhr);
      float z  = sigmoidf_(xv * wiz + biz + cz[i] + bhz);
      float nx = tanhf_(xv * win + bin + r * (cn[i] + bhn));
      float h2 = (1.0f - z) * nx + z * hp[i];
      hp[i] = h2;
      hn[m * NH + j] = (_Float16)h2;    // different buffer than reads -> safe
    }
    __syncthreads();                    // writes visible before next step reads
  }

  // ---- per-variable projection: concat[b0+m, v] = h_last[m,:] . Wp[v,:] + bp[v]
  #pragma unroll
  for (int i = 0; i < 8; ++i) h32[(sub * 8 + i) * NH + j] = hp[i];
  __syncthreads();

  if (tid < BTILE) {
    const float* wpv = Wp + v * NH;
    float acc = bp[v];
    for (int k = 0; k < NH; ++k) acc += h32[tid * NH + k] * wpv[k];
    concat[(b0 + tid) * NV + v] = acc;
  }
}

// ---------------------------------------------------------------------------
// FC head: [64,16] -> relu Linear(16,256) -> Linear(256,10). One block.
// ---------------------------------------------------------------------------
__global__ void __launch_bounds__(256)
fc_kernel(const float* __restrict__ concat,  // [B][V]
          const float* __restrict__ W1,      // [FC][V]
          const float* __restrict__ b1,      // [FC]
          const float* __restrict__ W2,      // [C][FC]
          const float* __restrict__ b2,      // [C]
          float* __restrict__ out)           // [B][C]
{
  extern __shared__ char smem2[];
  float* cs  = (float*)smem2;                        // [64][16]
  float* hfc = (float*)(smem2 + NB * NV * 4);        // [64][256]
  const int tid = threadIdx.x;

  for (int i = tid; i < NB * NV; i += 256) cs[i] = concat[i];
  __syncthreads();

  for (int i = tid; i < NB * NFC; i += 256) {
    int b = i / NFC, f = i % NFC;
    float acc = b1[f];
    #pragma unroll
    for (int k = 0; k < NV; ++k) acc += cs[b * NV + k] * W1[f * NV + k];
    hfc[i] = fmaxf(acc, 0.0f);
  }
  __syncthreads();

  for (int i = tid; i < NB * NC; i += 256) {
    int b = i / NC, c = i % NC;
    float acc = b2[c];
    for (int k = 0; k < NFC; ++k) acc += hfc[b * NFC + k] * W2[c * NFC + k];
    out[i] = acc;
  }
}

extern "C" void kernel_launch(void* const* d_in, const int* in_sizes, int n_in,
                              void* d_out, int out_size, void* d_ws, size_t ws_size,
                              hipStream_t stream) {
  const float* x   = (const float*)d_in[0];
  const float* Wih = (const float*)d_in[1];
  const float* Whh = (const float*)d_in[2];
  const float* bih = (const float*)d_in[3];
  const float* bhh = (const float*)d_in[4];
  const float* Wp  = (const float*)d_in[5];
  const float* bp  = (const float*)d_in[6];
  const float* W1  = (const float*)d_in[7];
  const float* b1  = (const float*)d_in[8];
  const float* W2  = (const float*)d_in[9];
  const float* b2  = (const float*)d_in[10];

  float* out    = (float*)d_out;
  float* concat = (float*)d_ws;   // [B][V] fp32, 4KB

  dim3 grid(NV, NB / BTILE);      // 16 heads x 4 batch tiles = 64 blocks
  gru_scan_kernel<<<grid, 256, 0, stream>>>(x, Wih, Whh, bih, bhh, Wp, bp,
                                            concat);

  size_t lds2 = (size_t)(NB * NV + NB * NFC) * sizeof(float);  // ~68KB
  fc_kernel<<<1, 256, lds2, stream>>>(concat, W1, b1, W2, b2, out);
}
